// RgnnLayer_34351148433957
// MI455X (gfx1250) — matrix operand order repaired and checked by reading, not compiled, not run on hardware
//
#include <hip/hip_runtime.h>

#define N_ENT    100000
#define EHALF    400000
#define DIM      128
#define OUT_MSG  (N_ENT * DIM)      // 12,800,000 floats
#define NUM_REL  400

typedef __attribute__((ext_vector_type(16))) __bf16 v16bf;
typedef __attribute__((ext_vector_type(8)))  __bf16 v8bf;
typedef __attribute__((ext_vector_type(8)))  float  v8f;

__device__ __forceinline__ __bf16 f2bf(float f) {
    unsigned u = __builtin_bit_cast(unsigned, f);
    u += 0x7fffu + ((u >> 16) & 1u);              // round-to-nearest-even
    unsigned short h = (unsigned short)(u >> 16);
    return __builtin_bit_cast(__bf16, h);
}

__device__ __forceinline__ void atomAddF(float* p, float v) {
    __hip_atomic_fetch_add(p, v, __ATOMIC_RELAXED, __HIP_MEMORY_SCOPE_AGENT);
}

// ---------------------------------------------------------------- zero
__global__ void zero_kernel(float* p, int n) {
    int i = blockIdx.x * blockDim.x + threadIdx.x;
    if (i < n) p[i] = 0.0f;
}

// ------------------------------------------------- pack weights to WMMA B layout
// fragment id = ((mat*8 + ntile)*4 + kstep)*32 + lane ; 16 bf16 per fragment
// lane: n = ntile*16 + (lane&15), K base = kstep*32 + (lane>>4)*16, elem j -> K=base+j
__global__ void pack_w_kernel(const float* w_in, const float* w_out,
                              const float* w_loop, __bf16* wp) {
    int t = blockIdx.x * blockDim.x + threadIdx.x;
    if (t >= 3 * 8 * 4 * 32) return;
    int lane = t & 31;
    int ks   = (t >> 5) & 3;
    int nt   = (t >> 7) & 7;
    int mat  = t >> 10;
    const float* w = (mat == 0) ? w_in : (mat == 1) ? w_out : w_loop;
    int n  = nt * 16 + (lane & 15);
    int kb = ks * 32 + (lane >> 4) * 16;
    __bf16* dst = wp + (size_t)t * 16;
#pragma unroll
    for (int j = 0; j < 16; ++j)
        dst[j] = f2bf(w[(kb + j) * DIM + n]);
}

// ---------------------------------------------------------------- degree count
__global__ void deg_kernel(const int* rowp, float* deg) {
    int e = blockIdx.x * blockDim.x + threadIdx.x;   // 0 .. 2*EHALF-1
    if (e < 2 * EHALF) {
        int row = rowp[e];
        atomAddF(deg + ((e < EHALF) ? 0 : N_ENT) + row, 1.0f);
    }
}

// ---------------------------------------------------------------- deg -> 1/sqrt
__global__ void dinv_kernel(float* deg, int n) {
    int i = blockIdx.x * blockDim.x + threadIdx.x;
    if (i < n) {
        float d = deg[i];
        deg[i] = (d > 0.0f) ? rsqrtf(d) : 0.0f;
    }
}

// ---------------------------------------------------------------- message kernel
// block = 16 edges; 8 waves, wave w owns N-tile w; 4 bf16 WMMAs (K=128)
__global__ void __launch_bounds__(256)
msg_kernel(const float* __restrict__ x, const float* __restrict__ r,
           const int* __restrict__ ei, const int* __restrict__ etp,
           const float* __restrict__ dinv_all,
           const __bf16* __restrict__ wp, float* __restrict__ out) {
    __shared__ __bf16 sA[16 * DIM];
    __shared__ int    sRow[16];
    __shared__ float  sNorm[16];

    int tid   = threadIdx.x;
    int g     = blockIdx.x;                 // 0..49999
    int half  = (g >= EHALF / 16) ? 1 : 0;
    int group = g - half * (EHALF / 16);
    int ebase = half * EHALF + group * 16;

    const int*   rowp = ei;
    const int*   colp = ei + 2 * EHALF;
    const float* dinv = dinv_all + half * N_ENT;

    // cooperative: sA[m][k] = bf16(x[col[m]][k] - r[etype[m]][k])
    int m     = tid >> 4;
    int kbase = (tid & 15) * 8;
    int e     = ebase + m;
    int col   = colp[e];
    int etv   = etp[e];
    const float4* xr = (const float4*)(x + col * DIM + kbase);
    const float4* rr = (const float4*)(r + etv * DIM + kbase);
    float4 x0 = xr[0], x1 = xr[1];
    float4 r0 = rr[0], r1 = rr[1];
    v8bf av;
    av[0] = f2bf(x0.x - r0.x); av[1] = f2bf(x0.y - r0.y);
    av[2] = f2bf(x0.z - r0.z); av[3] = f2bf(x0.w - r0.w);
    av[4] = f2bf(x1.x - r1.x); av[5] = f2bf(x1.y - r1.y);
    av[6] = f2bf(x1.z - r1.z); av[7] = f2bf(x1.w - r1.w);
    *(v8bf*)&sA[m * DIM + kbase] = av;
    if ((tid & 15) == 0) {
        int rw   = rowp[e];
        sRow[m]  = rw;
        sNorm[m] = dinv[rw] * dinv[col];
    }
    __syncthreads();

    int lane  = tid & 31;
    int ntile = tid >> 5;
    int h16   = lane >> 4;
    int mrow  = lane & 15;
    int kb    = h16 * 16;
    int nn    = lane & 15;

    const v16bf* bfrag = (const v16bf*)wp + ((size_t)(half * 8 + ntile) * 4) * 32;

    v8f c = {};
#pragma unroll
    for (int ks = 0; ks < 4; ++ks) {
        v8bf a0 = *(const v8bf*)&sA[mrow * DIM + ks * 32 + kb];
        v8bf a1 = *(const v8bf*)&sA[mrow * DIM + ks * 32 + kb + 8];
        v16bf a = __builtin_shufflevector(a0, a1, 0, 1, 2, 3, 4, 5, 6, 7,
                                                  8, 9, 10, 11, 12, 13, 14, 15);
        v16bf b = bfrag[ks * 32 + lane];
        c = __builtin_amdgcn_wmma_f32_16x16x32_bf16(false, a, false, b,
                                                    (short)0, c, false, false);
    }

#pragma unroll
    for (int v = 0; v < 8; ++v) {
        int me = v + 8 * h16;                        // edge index within group
        float val = c[v] * sNorm[me];
        atomAddF(&out[sRow[me] * DIM + ntile * 16 + nn], val);
    }
}

// ------------------------------------------- finalize: self-loop WMMA + tanh
__global__ void __launch_bounds__(256)
fin_kernel(const float* __restrict__ x, const float* __restrict__ loop_rel,
           const float* __restrict__ bias, const __bf16* __restrict__ wp,
           float* __restrict__ out) {
    __shared__ __bf16 sA[16 * DIM];
    int tid   = threadIdx.x;
    int mtile = blockIdx.x;                 // 0..6249

    int m     = tid >> 4;
    int kbase = (tid & 15) * 8;
    int row   = mtile * 16 + m;
    const float4* xr = (const float4*)(x + row * DIM + kbase);
    const float4* lr = (const float4*)(loop_rel + kbase);
    float4 x0 = xr[0], x1 = xr[1];
    float4 l0 = lr[0], l1 = lr[1];
    v8bf av;
    av[0] = f2bf(x0.x - l0.x); av[1] = f2bf(x0.y - l0.y);
    av[2] = f2bf(x0.z - l0.z); av[3] = f2bf(x0.w - l0.w);
    av[4] = f2bf(x1.x - l1.x); av[5] = f2bf(x1.y - l1.y);
    av[6] = f2bf(x1.z - l1.z); av[7] = f2bf(x1.w - l1.w);
    *(v8bf*)&sA[m * DIM + kbase] = av;
    __syncthreads();

    int lane  = tid & 31;
    int ntile = tid >> 5;
    int h16   = lane >> 4;
    int mrow  = lane & 15;
    int kb    = h16 * 16;
    int nn    = lane & 15;

    float bv = bias[ntile * 16 + nn];

    // C = accumulated in/out messages already in `out`
    v8f c;
#pragma unroll
    for (int v = 0; v < 8; ++v)
        c[v] = out[(mtile * 16 + v + 8 * h16) * DIM + ntile * 16 + nn];

    const v16bf* bfrag = (const v16bf*)wp + ((size_t)(2 * 8 + ntile) * 4) * 32;
#pragma unroll
    for (int ks = 0; ks < 4; ++ks) {
        v8bf a0 = *(const v8bf*)&sA[mrow * DIM + ks * 32 + kb];
        v8bf a1 = *(const v8bf*)&sA[mrow * DIM + ks * 32 + kb + 8];
        v16bf a = __builtin_shufflevector(a0, a1, 0, 1, 2, 3, 4, 5, 6, 7,
                                                  8, 9, 10, 11, 12, 13, 14, 15);
        v16bf b = bfrag[ks * 32 + lane];
        c = __builtin_amdgcn_wmma_f32_16x16x32_bf16(false, a, false, b,
                                                    (short)0, c, false, false);
    }

#pragma unroll
    for (int v = 0; v < 8; ++v) {
        int idx = (mtile * 16 + v + 8 * h16) * DIM + ntile * 16 + nn;
        out[idx] = tanhf(fmaf(c[v], (1.0f / 3.0f), bv));
    }
}

// ---------------------------------------------------------------- r @ w_rel
__global__ void rel_kernel(const float* __restrict__ r,
                           const float* __restrict__ w_rel,
                           float* __restrict__ r_out) {
    int t = blockIdx.x * blockDim.x + threadIdx.x;   // 400*128
    if (t >= NUM_REL * DIM) return;
    int i = t >> 7, j = t & 127;
    float acc = 0.0f;
#pragma unroll 8
    for (int k = 0; k < DIM; ++k)
        acc = fmaf(r[i * DIM + k], w_rel[k * DIM + j], acc);
    r_out[t] = acc;
}

// ================================================================ launcher
extern "C" void kernel_launch(void* const* d_in, const int* in_sizes, int n_in,
                              void* d_out, int out_size, void* d_ws, size_t ws_size,
                              hipStream_t stream) {
    const float* x        = (const float*)d_in[0];
    const float* r        = (const float*)d_in[1];
    const int*   ei       = (const int*)d_in[2];
    const int*   et       = (const int*)d_in[3];
    const float* w_in     = (const float*)d_in[4];
    const float* w_out    = (const float*)d_in[5];
    const float* w_loop   = (const float*)d_in[6];
    const float* w_rel    = (const float*)d_in[7];
    const float* loop_rel = (const float*)d_in[8];
    const float* bias     = (const float*)d_in[9];

    float* out   = (float*)d_out;           // 12,800,000 entity outputs
    float* r_out = out + OUT_MSG;           // 51,200 relation outputs

    float*  deg   = (float*)d_ws;                        // 2*N_ENT floats (deg -> dinv in place)
    __bf16* wpack = (__bf16*)((float*)d_ws + 2 * N_ENT); // 3*8*4*32*16 bf16 = 96 KB

    // 1. zero output accumulator + degree buffers
    zero_kernel<<<OUT_MSG / 256, 256, 0, stream>>>(out, OUT_MSG);
    zero_kernel<<<(2 * N_ENT + 255) / 256, 256, 0, stream>>>(deg, 2 * N_ENT);

    // 2. pack weights into WMMA B-fragment layout (bf16)
    pack_w_kernel<<<(3 * 8 * 4 * 32 + 255) / 256, 256, 0, stream>>>(w_in, w_out, w_loop, wpack);

    // 3. degree counts (per half), 4. deg -> 1/sqrt(deg)
    deg_kernel<<<(2 * EHALF) / 256, 256, 0, stream>>>(ei, deg);
    dinv_kernel<<<(2 * N_ENT + 255) / 256, 256, 0, stream>>>(deg, 2 * N_ENT);

    // 5. fused gather -> bf16 WMMA -> norm-scale -> atomic scatter (both halves)
    msg_kernel<<<2 * (EHALF / 16), 256, 0, stream>>>(x, r, ei, et, deg, wpack, out);

    // 6. self-loop WMMA + (sum/3 + bias) -> tanh
    fin_kernel<<<N_ENT / 16, 256, 0, stream>>>(x, loop_rel, bias, wpack, out);

    // 7. relation transform
    rel_kernel<<<(NUM_REL * DIM + 255) / 256, 256, 0, stream>>>(r, w_rel, r_out);
}
